// SuperposedExpert_82497731821705
// MI455X (gfx1250) — compile-verified, matrix operand order checked by reading, not run on hardware
//
#include <hip/hip_runtime.h>

// ---------------------------------------------------------------------------
// SuperposedExpert fused CDNA5 kernel (gfx1250, wave32, WMMA bf16 16x16x32)
// Round 3: +async global->LDS staging (ASYNCcnt path), non-temporal output
//          stores. Keeps padded LDS, cached A-tiles, SE-scope prefetches.
// ---------------------------------------------------------------------------

typedef __bf16 bf16_t;
typedef __attribute__((ext_vector_type(16))) __bf16 v16bf;
typedef __attribute__((ext_vector_type(8)))  __bf16 v8bf;
typedef __attribute__((ext_vector_type(8)))  float  v8f;

#define BATCH   8192
#define DMODEL  1024
#define HD_DIM  512
#define KPATH   4
#define BT      16        // batch rows per workgroup

// ---- workspace byte offsets (all 256B aligned) ----
#define OFF_XBF    0ull                    // [8192][1024] bf16, cols permuted d' = p*32+m
#define OFF_W1AT   16777216ull             // [K][64 n1][16 r][32 m]   = c1a^T
#define OFF_W1BT   17039360ull             // [K][64 o][512 (p*16+r)]  = c1b^T
#define OFF_W2AT   17301504ull             // [K][32 n2][16 r][64 m2]  = c2a^T
#define OFF_W2BT   17563648ull             // [K][32 pp2][32 o2][32 (pp*16+r)] = c2b^T
#define OFF_HDINT  17825792ull             // [512 h][1024 d]          = hd_in^T
#define OFF_HDINPT 18874368ull             // [512 h][1024 d']         = hd_in^T, cols permuted
#define OFF_HDOUTT 19922944ull             // [1024 n][512 h]          = hd_out^T

// ---- padded LDS row strides (elements): +8 elems => row stride = 4 dwords
// mod 64 banks, so the 16 rows of a WMMA operand tile cover all 64 banks ----
#define XT_LD   1032      // xt   [16][1024 -> 1032]
#define HB_LD   4104      // hbuf [16][4096 -> 4104]
#define T1_LD    520      // t1   [2][16][512 -> 520]
#define T1_SZ   8320      // 16 * 520
#define T2_LD   1032      // t2p  [16][1024 -> 1032]
#define YB_LD   1032      // ybuf [16][1024 -> 1032]
#define CB_LD    520      // cbuf [16][512 -> 520]
#define XHD_LD   516      // xhd  [16][512 -> 516] (f32)

// ---- LDS byte offsets ----
#define L_XT    0          // xt: 16*1032*2                      = 33024 B
#define L_BIG   33024      // hbuf 16*4104*2=131328 | xhd 16*516*4 | ybuf 16*1032*2
#define L_T1    164352     // t1 2*8320*2=33280 | t2p 16*1032*2 | cbuf 16*520*2
#define L_SC    197632     // scores [16][4] f32
#define L_PR    197888     // probs  [16][4] f32
#define SMEM_BYTES 198144  // ~193.5 KB of the 320 KB/WGP LDS

__device__ __forceinline__ v8f wmma_bf16(v16bf a, v16bf b, v8f c) {
  return __builtin_amdgcn_wmma_f32_16x16x32_bf16(false, a, false, b, (short)0, c,
                                                 false, false);
}

// Load one WMMA operand (A from row-major [16][ld], or B from B^T row-major
// [16 N][ld]).  Lane l owns row/col l%16; its 16 halves map to
// K = e + 8*(e>=8) + 8*(l/16), i.e. two contiguous 16-byte chunks.
__device__ __forceinline__ v16bf ld_op(const bf16_t* __restrict__ base, int ld, int lane) {
  const int row = lane & 15;
  const int hi  = lane >> 4;
  const bf16_t* p = base + row * ld + hi * 8;
  union { v16bf v; v8bf h[2]; } u;
  u.h[0] = *(const v8bf*)(p);        // K = 8*hi   .. 8*hi+7
  u.h[1] = *(const v8bf*)(p + 16);   // K = 16+8*hi .. 16+8*hi+7
  return u.v;
}

// Store a 16x16 f32 C/D tile (row = v + 8*(l/16), col = l%16) as bf16.
__device__ __forceinline__ void st_cd_bf16(bf16_t* base, int ld, int lane, v8f c) {
  const int col = lane & 15;
  const int r0  = (lane >> 4) * 8;
#pragma unroll
  for (int v = 0; v < 8; ++v) base[(r0 + v) * ld + col] = (bf16_t)c[v];
}

__device__ __forceinline__ float gelu_tanh(float x) {
  float x3 = x * x * x;
  return 0.5f * x * (1.0f + tanhf(0.7978845608028654f * (x + 0.044715f * x3)));
}

// ---------------------------------------------------------------------------
// Kernel 0: bf16 conversion + layout packing of x and all weights into d_ws.
// ---------------------------------------------------------------------------
__global__ __launch_bounds__(256) void sp_prepack(
    const float* __restrict__ x,
    const float* __restrict__ c1a, const float* __restrict__ c1b,
    const float* __restrict__ c2a, const float* __restrict__ c2b,
    const float* __restrict__ hd_in, const float* __restrict__ hd_out,
    bf16_t* __restrict__ xbfP,
    bf16_t* __restrict__ W1aT, bf16_t* __restrict__ W1bT,
    bf16_t* __restrict__ W2aT, bf16_t* __restrict__ W2bT,
    bf16_t* __restrict__ hdInT, bf16_t* __restrict__ hdInPT,
    bf16_t* __restrict__ hdOutT)
{
  const long NX = (long)BATCH * DMODEL;   // 8388608
  const long NW = 131072;                 // each TT core pack
  const long NH = 524288;                 // each HD matrix pack
  const long total = NX + 4 * NW + 3 * NH;
  for (long i = (long)blockIdx.x * blockDim.x + threadIdx.x; i < total;
       i += (long)gridDim.x * blockDim.x) {
    long j = i;
    if (j < NX) {                               // x, columns permuted d' = p*32+m
      long b = j >> 10; int t = (int)(j & 1023); int p = t >> 5; int m = t & 31;
      xbfP[j] = (bf16_t)x[(b << 10) + m * 32 + p];
    } else if ((j -= NX) < NW) {                // W1aT[k][n1][r][m] = c1a[k][m][n1][r]
      int m = (int)(j & 31), r = (int)((j >> 5) & 15), n1 = (int)((j >> 9) & 63),
          k = (int)(j >> 15);
      W1aT[j] = (bf16_t)c1a[(((k * 32 + m) * 64 + n1) << 4) + r];
    } else if ((j -= NW) < NW) {                // W1bT[k][o][p*16+r] = c1b[k][r][p][o]
      int r = (int)(j & 15), p = (int)((j >> 4) & 31), o = (int)((j >> 9) & 63),
          k = (int)(j >> 15);
      W1bT[j] = (bf16_t)c1b[((long)((k * 16 + r) * 32 + p) << 6) + o];
    } else if ((j -= NW) < NW) {                // W2aT[k][n2][r][m2] = c2a[k][m2][n2][r]
      int m2 = (int)(j & 63), r = (int)((j >> 6) & 15), n2 = (int)((j >> 10) & 31),
          k = (int)(j >> 15);
      W2aT[j] = (bf16_t)c2a[(((k * 64 + m2) * 32 + n2) << 4) + r];
    } else if ((j -= NW) < NW) {                // W2bT[k][pp2][o2][pp*16+r] = c2b[k][r][2pp2+pp][o2]
      int r = (int)(j & 15), pp = (int)((j >> 4) & 1), o2 = (int)((j >> 5) & 31),
          pp2 = (int)((j >> 10) & 31), k = (int)(j >> 15);
      W2bT[j] = (bf16_t)c2b[((long)((k * 16 + r) * 64 + (pp2 * 2 + pp)) << 5) + o2];
    } else if ((j -= NW) < NH) {                // hdInT[h][d] = hd_in[d][h]
      int d = (int)(j & 1023); int h = (int)(j >> 10);
      hdInT[j] = (bf16_t)hd_in[((long)d << 9) + h];
    } else if ((j -= NH) < NH) {                // hdInPT[h][p*32+m] = hd_in[m*32+p][h]
      int t = (int)(j & 1023); int p = t >> 5, m = t & 31; int h = (int)(j >> 10);
      hdInPT[j] = (bf16_t)hd_in[((long)(m * 32 + p) << 9) + h];
    } else {                                    // hdOutT[n][h] = hd_out[h][n]
      j -= NH;
      int h = (int)(j & 511); int n = (int)(j >> 9);
      hdOutT[j] = (bf16_t)hd_out[((long)h << 10) + n];
    }
  }
}

// ---------------------------------------------------------------------------
// Kernel 1: fully fused per-tile pipeline (routing -> K x (TT1+gelu -> TT2 ->
// HD-bind/collapse) -> hd_out projection), everything via WMMA bf16.
// ---------------------------------------------------------------------------
__global__ __launch_bounds__(256) void sp_main(
    const bf16_t* __restrict__ xbfP,
    const bf16_t* __restrict__ W1aT, const bf16_t* __restrict__ W1bT,
    const bf16_t* __restrict__ W2aT, const bf16_t* __restrict__ W2bT,
    const bf16_t* __restrict__ hdInT, const bf16_t* __restrict__ hdInPT,
    const bf16_t* __restrict__ hdOutT,
    const float* __restrict__ path_bases, const float* __restrict__ path_weights,
    float* __restrict__ out)
{
  extern __shared__ char smem[];
  bf16_t* xt   = (bf16_t*)(smem + L_XT);
  bf16_t* hbuf = (bf16_t*)(smem + L_BIG);   // [b][p2][m2]  (b*HB_LD + p2*64 + m2)
  float*  xhd  = (float* )(smem + L_BIG);   // alias (used before hbuf)
  bf16_t* ybuf = (bf16_t*)(smem + L_BIG);   // alias (used after hbuf consumed)
  bf16_t* t1   = (bf16_t*)(smem + L_T1);    // [2][16][T1_LD]
  bf16_t* t2p  = (bf16_t*)(smem + L_T1);    // alias [16][T2_LD]
  bf16_t* cbuf = (bf16_t*)(smem + L_T1);    // alias [16][CB_LD]
  float*  scoresL = (float*)(smem + L_SC);
  float*  probsL  = (float*)(smem + L_PR);

  const int tid  = threadIdx.x;
  const int lane = tid & 31;                 // wave32
  const int wave = tid >> 5;                 // 8 waves
  const int col  = lane & 15;
  const int r0   = (lane >> 4) * 8;
  const int b0   = blockIdx.x * BT;
  const v8f vzero = {0.f, 0.f, 0.f, 0.f, 0.f, 0.f, 0.f, 0.f};

  // -------- Ph0: async-stage x tile into padded LDS (ASYNCcnt path) -------
  // GLOBAL_LOAD_ASYNC_TO_LDS_B128 copies 16B/lane directly to LDS without a
  // VGPR round-trip.  The VDST operand is the LDS byte address, which equals
  // the low 32 bits of the generic pointer (ISA 10.2: LDS aperture addresses
  // truncate to addr[31:0]).
  {
    const bf16_t* xg = xbfP + (size_t)b0 * DMODEL;
    for (int i = tid; i < BT * DMODEL / 8; i += 256) {
      const int row = i >> 7, c8 = i & 127;          // 128 chunks of 8 per row
      const unsigned lds_addr =
          (unsigned)(unsigned long long)(uintptr_t)(xt + row * XT_LD + c8 * 8);
      const unsigned long long gaddr =
          (unsigned long long)(uintptr_t)(xg + row * DMODEL + c8 * 8);
      asm volatile("global_load_async_to_lds_b128 %0, %1, off"
                   :: "v"(lds_addr), "v"(gaddr) : "memory");
    }
    asm volatile("s_wait_asynccnt 0" ::: "memory");
  }
  __syncthreads();

  // -------- Ph1: x_hd = x @ hd_in  ([16x1024] @ [1024x512]) --------
  for (int t = 0; t < 4; ++t) {
    const int nt = wave * 4 + t;
    v8f c = vzero;
    for (int kc = 0; kc < 32; ++kc) {
      if (kc + 4 < 32)
        __builtin_prefetch(hdInPT + (size_t)nt * 16 * 1024 + (kc + 4) * 32, 0, 3);
      v16bf a = ld_op(xt + kc * 32, XT_LD, lane);
      v16bf b = ld_op(hdInPT + (size_t)nt * 16 * 1024 + kc * 32, 1024, lane);
      c = wmma_bf16(a, b, c);
    }
#pragma unroll
    for (int v = 0; v < 8; ++v) xhd[(r0 + v) * XHD_LD + nt * 16 + col] = c[v];
  }
  __syncthreads();

  // -------- Ph2: scores + softmax over K --------
  if (tid < BT * KPATH) {
    const int b = tid >> 2, k = tid & 3;
    float s = 0.f;
    for (int h = 0; h < HD_DIM; ++h)
      s += xhd[b * XHD_LD + h] * path_bases[k * HD_DIM + h];
    scoresL[tid] = s;   // TEMP = 1.0
  }
  __syncthreads();
  if (tid < BT) {
    float m = scoresL[tid * 4];
    for (int k = 1; k < 4; ++k) m = fmaxf(m, scoresL[tid * 4 + k]);
    float e[4], s = 0.f;
    for (int k = 0; k < 4; ++k) { e[k] = __expf(scoresL[tid * 4 + k] - m); s += e[k]; }
    for (int k = 0; k < 4; ++k) probsL[tid * 4 + k] = e[k] / s;
  }
  __syncthreads();

  // Register-cache the wave's 8 TT1 core-A x-tiles (invariant over n1p and k).
  v16bf xA[8];
#pragma unroll
  for (int j = 0; j < 8; ++j) {
    const int p1 = (wave * 8 + j) & 31;
    xA[j] = ld_op(xt + p1 * 32, XT_LD, lane);
  }

  v8f colAcc[4] = {vzero, vzero, vzero, vzero};   // persistent collapsed[b][hd] tiles
  v8f yacc[8];                                    // persistent y[b][1024] tiles (per k)

  for (int k = 0; k < KPATH; ++k) {
    // ================= TT1: h = gelu(TT(x; c1a, c1b)) =================
    for (int n1p = 0; n1p < 32; ++n1p) {
      // core A: t1[(b,p1), r1] = sum_m1 x[b,m1,p1] * c1a[k,m1,n1,r1]
#pragma unroll
      for (int j = 0; j < 8; ++j) {
        const int job = wave * 8 + j;           // 64 jobs = 2 n1 x 32 p1
        const int nb  = job >> 5;
        const int p1  = job & 31;
        const int n1  = n1p * 2 + nb;
        v16bf b = ld_op(W1aT + (size_t)((k * 64 + n1) * 16) * 32, 32, lane);
        v8f c = wmma_bf16(xA[j], b, vzero);
        st_cd_bf16(t1 + nb * T1_SZ + p1 * 16, T1_LD, lane, c);
      }
      __syncthreads();
      // core B: h[b,n1,o1] = sum_{p1,r1} t1[b,(p1,r1)] * c1b[k,r1,p1,o1]; gelu
      {
        const int nb = wave >> 2, ot = wave & 3;   // 8 jobs = 2 n1 x 4 o-tiles
        const int n1 = n1p * 2 + nb;
        v8f c = vzero;
        for (int kc = 0; kc < 16; ++kc) {
          v16bf a = ld_op(t1 + nb * T1_SZ + kc * 32, T1_LD, lane);
          v16bf b = ld_op(W1bT + (size_t)(k * 64 + ot * 16) * 512 + kc * 32, 512, lane);
          c = wmma_bf16(a, b, c);
        }
#pragma unroll
        for (int v = 0; v < 8; ++v) {
          float h = gelu_tanh(c[v]);
          // scatter: hbuf[b][p2 = o1][m2 = n1]
          hbuf[(r0 + v) * HB_LD + (ot * 16 + col) * 64 + n1] = (bf16_t)h;
        }
      }
      __syncthreads();
    }

    // ================= TT2: y = TT(h; c2a, c2b) =================
#pragma unroll
    for (int j = 0; j < 8; ++j) yacc[j] = vzero;
    for (int pp2 = 0; pp2 < 32; ++pp2) {
      const int p2 = 2 * pp2 + (wave & 1);
#pragma unroll
      for (int jj = 0; jj < 8; ++jj) {
        const int n2 = (wave >> 1) + 4 * jj;
        v8f c = vzero;
        for (int kc = 0; kc < 2; ++kc) {
          v16bf a = ld_op(hbuf + p2 * 64 + kc * 32, HB_LD, lane);
          v16bf b = ld_op(W2aT + (size_t)((k * 32 + n2) * 16) * 64 + kc * 32, 64, lane);
          c = wmma_bf16(a, b, c);
        }
        st_cd_bf16(t2p + n2 * 32 + (p2 & 1) * 16, T2_LD, lane, c);
      }
      __syncthreads();
#pragma unroll
      for (int jj = 0; jj < 8; ++jj) {
        const int n2 = (wave >> 1) + 4 * jj;
        v16bf a = ld_op(t2p + n2 * 32, T2_LD, lane);
        v16bf b = ld_op(W2bT + (size_t)((k * 32 + pp2) * 32 + (wave & 1) * 16) * 32,
                        32, lane);
        yacc[jj] = wmma_bf16(a, b, yacc[jj]);   // accumulate over p2 pairs
      }
      __syncthreads();
    }

    // dump y (bf16) to LDS for the HD projection
#pragma unroll
    for (int jj = 0; jj < 8; ++jj) {
      const int n2 = (wave >> 1) + 4 * jj;
      st_cd_bf16(ybuf + n2 * 32 + (wave & 1) * 16, YB_LD, lane, yacc[jj]);
    }
    __syncthreads();

    // ============ y_hd = y @ hd_in ; bind + softmax collapse ============
    for (int t = 0; t < 4; ++t) {
      const int nt = wave * 4 + t;
      v8f c = vzero;
      for (int kc = 0; kc < 32; ++kc) {
        if (kc + 4 < 32)
          __builtin_prefetch(hdInT + (size_t)nt * 16 * 1024 + (kc + 4) * 32, 0, 3);
        v16bf a = ld_op(ybuf + kc * 32, YB_LD, lane);
        v16bf b = ld_op(hdInT + (size_t)nt * 16 * 1024 + kc * 32, 1024, lane);
        c = wmma_bf16(a, b, c);
      }
      const float w_hd = path_weights[k * HD_DIM + nt * 16 + col];
#pragma unroll
      for (int v = 0; v < 8; ++v)
        colAcc[t][v] += probsL[(r0 + v) * 4 + k] * w_hd * c[v];
    }
    __syncthreads();   // ybuf/hbuf region reused next k
  }

  // -------- out = collapsed @ hd_out  ([16x512] @ [512x1024]) --------
  for (int t = 0; t < 4; ++t)
    st_cd_bf16(cbuf + (wave * 4 + t) * 16, CB_LD, lane, colAcc[t]);
  __syncthreads();
  for (int t = 0; t < 8; ++t) {
    const int nt = wave * 8 + t;
    v8f c = vzero;
    for (int kc = 0; kc < 16; ++kc) {
      v16bf a = ld_op(cbuf + kc * 32, CB_LD, lane);
      v16bf b = ld_op(hdOutT + (size_t)nt * 16 * 512 + kc * 32, 512, lane);
      c = wmma_bf16(a, b, c);
    }
    // write-once output: non-temporal so it doesn't evict L2-resident weights
#pragma unroll
    for (int v = 0; v < 8; ++v)
      __builtin_nontemporal_store(
          c[v], &out[(size_t)(b0 + r0 + v) * DMODEL + nt * 16 + col]);
  }
}

// ---------------------------------------------------------------------------
extern "C" void kernel_launch(void* const* d_in, const int* in_sizes, int n_in,
                              void* d_out, int out_size, void* d_ws, size_t ws_size,
                              hipStream_t stream) {
  (void)in_sizes; (void)n_in; (void)out_size; (void)ws_size;
  const float* x   = (const float*)d_in[0];
  const float* c1a = (const float*)d_in[1];
  const float* c1b = (const float*)d_in[2];
  const float* c2a = (const float*)d_in[3];
  const float* c2b = (const float*)d_in[4];
  const float* hdi = (const float*)d_in[5];
  const float* hdo = (const float*)d_in[6];
  const float* pb  = (const float*)d_in[7];
  const float* pw  = (const float*)d_in[8];
  float* out = (float*)d_out;
  char*  ws  = (char*)d_ws;

  bf16_t* xbfP   = (bf16_t*)(ws + OFF_XBF);
  bf16_t* W1aTp  = (bf16_t*)(ws + OFF_W1AT);
  bf16_t* W1bTp  = (bf16_t*)(ws + OFF_W1BT);
  bf16_t* W2aTp  = (bf16_t*)(ws + OFF_W2AT);
  bf16_t* W2bTp  = (bf16_t*)(ws + OFF_W2BT);
  bf16_t* hdInTp = (bf16_t*)(ws + OFF_HDINT);
  bf16_t* hdInPTp= (bf16_t*)(ws + OFF_HDINPT);
  bf16_t* hdOutTp= (bf16_t*)(ws + OFF_HDOUTT);

  sp_prepack<<<4096, 256, 0, stream>>>(x, c1a, c1b, c2a, c2b, hdi, hdo,
      xbfP, W1aTp, W1bTp, W2aTp, W2bTp, hdInTp, hdInPTp, hdOutTp);

  (void)hipFuncSetAttribute((const void*)sp_main,
      hipFuncAttributeMaxDynamicSharedMemorySize, SMEM_BYTES);
  sp_main<<<BATCH / BT, 256, SMEM_BYTES, stream>>>(
      xbfP, W1aTp, W1bTp, W2aTp, W2bTp, hdInTp, hdInPTp, hdOutTp, pb, pw, out);
}
